// ProTCL_61778809585686
// MI455X (gfx1250) — compile-verified, matrix operand order checked by reading, not run on hardware
//
#include <hip/hip_runtime.h>
#include <hip/hip_bf16.h>

typedef __bf16 bf16;
typedef __attribute__((ext_vector_type(4)))  float  v4f;
typedef __attribute__((ext_vector_type(8)))  float  v8f;
typedef __attribute__((ext_vector_type(8)))  __bf16 v8bf;
typedef __attribute__((ext_vector_type(16))) __bf16 v16bf;

#define B_  8
#define L_  4000
#define PD_ 1100
#define LD_ 1024
#define D_  1024
#define H_  2048

static __device__ __forceinline__ v16bf cat16(v8bf lo, v8bf hi) {
  return __builtin_shufflevector(lo, hi, 0,1,2,3,4,5,6,7,8,9,10,11,12,13,14,15);
}

// ---------------------------------------------------------------------------
// f32 -> bf16 (with optional column slice of a strided matrix, for W1l)
// ---------------------------------------------------------------------------
__global__ void cvt_bf16(const float* __restrict__ src, bf16* __restrict__ dst,
                         int rows, int cols, int src_stride, int src_off) {
  int n = rows * cols;
  for (int i = blockIdx.x * blockDim.x + threadIdx.x; i < n;
       i += gridDim.x * blockDim.x) {
    int r = i / cols, c = i - r * cols;
    dst[i] = (bf16)src[(size_t)r * src_stride + src_off + c];
  }
}

// ---------------------------------------------------------------------------
// P-path (tiny, keep full f32): P_e = seq @ Wp.T ; Pw1 = P_e @ W1p.T + b1
// ---------------------------------------------------------------------------
__global__ void pe_kernel(const float* __restrict__ seq,
                          const float* __restrict__ Wp,
                          float* __restrict__ Pe) {
  int i = blockIdx.x * blockDim.x + threadIdx.x;
  if (i >= B_ * D_) return;
  int bi = i / D_, j = i - bi * D_;
  const float* s = seq + bi * PD_;
  const float* w = Wp + (size_t)j * PD_;
  float acc = 0.f;
  for (int k = 0; k < PD_; ++k) acc = fmaf(s[k], w[k], acc);
  Pe[i] = acc;
}

__global__ void pw1_kernel(const float* __restrict__ Pe,
                           const float* __restrict__ W1,
                           const float* __restrict__ b1,
                           float* __restrict__ Pw1) {
  int i = blockIdx.x * blockDim.x + threadIdx.x;
  if (i >= B_ * H_) return;
  int bi = i / H_, j = i - bi * H_;
  const float* p = Pe + bi * D_;
  const float* w = W1 + (size_t)j * (2 * D_);   // W1p = W1[:, :D]
  float acc = b1[j];
  for (int k = 0; k < D_; ++k) acc = fmaf(p[k], w[k], acc);
  Pw1[i] = acc;
}

// ---------------------------------------------------------------------------
// C(MxN) = A(MxK) . B(NxK)^T   bf16 WMMA, f32 accum.
// Wave tile 16x64 (4 accumulators), 8 waves/block cover 512 N.
// Requires: M%16==0, N%512==0, K%32==0.
// ---------------------------------------------------------------------------
template <bool OUT_BF16>
__global__ __launch_bounds__(256) void gemm_bf16_nt(
    const bf16* __restrict__ A, const bf16* __restrict__ Bm,
    float* __restrict__ Cf, bf16* __restrict__ Cb, int N, int K) {
  const int lane = threadIdx.x & 31;
  const int wave = threadIdx.x >> 5;
  const int r    = lane & 15;
  const int hi   = lane >> 4;
  const int m0   = blockIdx.y * 16;
  const int n0   = (blockIdx.x * 8 + wave) * 64;

  const bf16* arow = A + (size_t)(m0 + r) * K;
  v8f c[4];
#pragma unroll
  for (int t = 0; t < 4; ++t) c[t] = (v8f){};

  for (int k0 = 0; k0 < K; k0 += 32) {
    v8bf alo = *(const v8bf*)(arow + k0 + hi * 8);
    v8bf ahi = *(const v8bf*)(arow + k0 + 16 + hi * 8);
    v16bf a  = cat16(alo, ahi);
#pragma unroll
    for (int t = 0; t < 4; ++t) {
      const bf16* brow = Bm + (size_t)(n0 + t * 16 + r) * K;
      v8bf blo = *(const v8bf*)(brow + k0 + hi * 8);
      v8bf bhi = *(const v8bf*)(brow + k0 + 16 + hi * 8);
      c[t] = __builtin_amdgcn_wmma_f32_16x16x32_bf16(
          false, a, false, cat16(blo, bhi), (short)0, c[t], false, false);
    }
  }
#pragma unroll
  for (int t = 0; t < 4; ++t) {
    int col = n0 + t * 16 + r;
#pragma unroll
    for (int v = 0; v < 8; ++v) {
      int row = m0 + hi * 8 + v;
      if (OUT_BF16) Cb[(size_t)row * N + col] = (bf16)c[t][v];
      else          Cf[(size_t)row * N + col] = c[t][v];
    }
  }
}

// ---------------------------------------------------------------------------
// Fused dominant stage: per block = 32 (b,l) rows.
//   h1 = relu(Pw1[b] + Lw1[l])        -> bf16 in LDS (32x2048, xor-swizzled)
//   y  = h1 @ W2.T  (WMMA sweep over all 128 N-tiles)
//   logit = sum_g relu(y+b2[g]) * Wout[g] + bout   (LDS ds_add_f32 reduce)
// Inner loop is 2x2 register-blocked (2 M-subtiles x 2 N-tiles) so each
// ds_load_b128 A-fragment feeds 2 WMMAs: 1 LDS load + 1 global load per WMMA.
// ---------------------------------------------------------------------------
__global__ __launch_bounds__(256) void fused_out(
    const float* __restrict__ Pw1, const float* __restrict__ Lw1,
    const bf16* __restrict__ W2b, const float* __restrict__ b2,
    const float* __restrict__ Wout, const float* __restrict__ bout,
    float* __restrict__ out) {
  extern __shared__ char smem[];
  bf16* h1s = (bf16*)smem;                    // 32 * 2048 bf16 = 128 KB

  const int tile = blockIdx.x;                // 0..999
  const int bi   = tile / 125;
  const int l0   = (tile - bi * 125) * 32;
  const int tid  = threadIdx.x;

  // ---- phase 1: build bf16 h1 tile in LDS (16B-chunk xor swizzle) ----
  const float* prow = Pw1 + bi * H_;
#pragma unroll 4
  for (int it = 0; it < 32; ++it) {
    int cid = tid + it * 256;                 // 0..8191
    int row = cid >> 8;                       // 0..31
    int ch  = cid & 255;                      // 8-element chunk
    const float* lp = Lw1 + (size_t)(l0 + row) * H_ + ch * 8;
    const float* pp = prow + ch * 8;
    v4f la0 = *(const v4f*)(lp);
    v4f la1 = *(const v4f*)(lp + 4);
    v4f pa0 = *(const v4f*)(pp);
    v4f pa1 = *(const v4f*)(pp + 4);
    v8bf pk;
#pragma unroll
    for (int j = 0; j < 4; ++j) {
      float f0 = pa0[j] + la0[j];
      float f1 = pa1[j] + la1[j];
      pk[j]     = (bf16)(f0 > 0.f ? f0 : 0.f);
      pk[4 + j] = (bf16)(f1 > 0.f ? f1 : 0.f);
    }
    *(v8bf*)(h1s + row * H_ + ((ch ^ (row & 15)) * 8)) = pk;
  }
  __syncthreads();

  // ---- phase 2: WMMA sweep; each wave owns 16 N-tiles, processed in pairs --
  const int lane = tid & 31;
  const int wave = tid >> 5;
  const int r    = lane & 15;
  const int hi   = lane >> 4;

  float acc0[8], acc1[8];
#pragma unroll
  for (int v = 0; v < 8; ++v) { acc0[v] = 0.f; acc1[v] = 0.f; }

  for (int p = 0; p < 8; ++p) {
    const int g0 = (wave * 16 + 2 * p) * 16 + r;   // output columns
    const int g1 = g0 + 16;
    const bf16* w0 = W2b + (size_t)g0 * H_;
    const bf16* w1 = W2b + (size_t)g1 * H_;
    v8f c00 = (v8f){};   // m-sub0 x n0
    v8f c10 = (v8f){};   // m-sub1 x n0
    v8f c01 = (v8f){};   // m-sub0 x n1
    v8f c11 = (v8f){};   // m-sub1 x n1
    for (int k0 = 0; k0 < H_; k0 += 32) {
      int ch0 = (k0 >> 3) + hi;               // chunk of low 16 halves
      int ch1 = ch0 + 2;                      // chunk of high 16 halves
      v8bf a0lo = *(const v8bf*)(h1s + r * H_        + ((ch0 ^ r) * 8));
      v8bf a0hi = *(const v8bf*)(h1s + r * H_        + ((ch1 ^ r) * 8));
      v8bf a1lo = *(const v8bf*)(h1s + (16 + r) * H_ + ((ch0 ^ r) * 8));
      v8bf a1hi = *(const v8bf*)(h1s + (16 + r) * H_ + ((ch1 ^ r) * 8));
      v16bf a0 = cat16(a0lo, a0hi);
      v16bf a1 = cat16(a1lo, a1hi);
      v8bf b0lo = *(const v8bf*)(w0 + k0 + hi * 8);
      v8bf b0hi = *(const v8bf*)(w0 + k0 + 16 + hi * 8);
      v8bf b1lo = *(const v8bf*)(w1 + k0 + hi * 8);
      v8bf b1hi = *(const v8bf*)(w1 + k0 + 16 + hi * 8);
      v16bf bb0 = cat16(b0lo, b0hi);
      v16bf bb1 = cat16(b1lo, b1hi);
      c00 = __builtin_amdgcn_wmma_f32_16x16x32_bf16(
          false, a0, false, bb0, (short)0, c00, false, false);
      c10 = __builtin_amdgcn_wmma_f32_16x16x32_bf16(
          false, a1, false, bb0, (short)0, c10, false, false);
      c01 = __builtin_amdgcn_wmma_f32_16x16x32_bf16(
          false, a0, false, bb1, (short)0, c01, false, false);
      c11 = __builtin_amdgcn_wmma_f32_16x16x32_bf16(
          false, a1, false, bb1, (short)0, c11, false, false);
    }
    const float bias0 = b2[g0];
    const float wo0   = Wout[g0];
    const float bias1 = b2[g1];
    const float wo1   = Wout[g1];
#pragma unroll
    for (int v = 0; v < 8; ++v) {
      float y00 = c00[v] + bias0;
      float y01 = c01[v] + bias1;
      float y10 = c10[v] + bias0;
      float y11 = c11[v] + bias1;
      acc0[v] += (y00 > 0.f ? y00 : 0.f) * wo0 + (y01 > 0.f ? y01 : 0.f) * wo1;
      acc1[v] += (y10 > 0.f ? y10 : 0.f) * wo0 + (y11 > 0.f ? y11 : 0.f) * wo1;
    }
  }

  // ---- phase 3: reduce 32 logits via LDS atomics (smem reused) ----
  __syncthreads();
  float* partial = (float*)smem;
  if (tid < 32) partial[tid] = 0.f;
  __syncthreads();
#pragma unroll
  for (int v = 0; v < 8; ++v) {
    atomicAdd(&partial[hi * 8 + v],      acc0[v]);
    atomicAdd(&partial[16 + hi * 8 + v], acc1[v]);
  }
  __syncthreads();
  if (tid < 32) out[(size_t)bi * L_ + l0 + tid] = partial[tid] + bout[0];
}

// ---------------------------------------------------------------------------
extern "C" void kernel_launch(void* const* d_in, const int* in_sizes, int n_in,
                              void* d_out, int out_size, void* d_ws, size_t ws_size,
                              hipStream_t stream) {
  (void)in_sizes; (void)n_in; (void)out_size; (void)ws_size;
  const float* seq  = (const float*)d_in[0];
  const float* lab  = (const float*)d_in[1];
  const float* Wp   = (const float*)d_in[2];
  const float* Wl   = (const float*)d_in[3];
  const float* W1   = (const float*)d_in[4];
  const float* b1   = (const float*)d_in[5];
  const float* W2   = (const float*)d_in[6];
  const float* b2   = (const float*)d_in[7];
  const float* Wout = (const float*)d_in[8];
  const float* bout = (const float*)d_in[9];
  float* out = (float*)d_out;

  char* ws = (char*)d_ws;
  size_t off = 0;
  auto alloc = [&](size_t bytes) {
    char* p = ws + off;
    off += (bytes + 255) & ~(size_t)255;
    return p;
  };
  bf16*  W2b  = (bf16*)alloc((size_t)H_ * H_ * 2);       // 8 MB
  bf16*  Wlb  = (bf16*)alloc((size_t)D_ * LD_ * 2);      // 2 MB
  bf16*  W1lb = (bf16*)alloc((size_t)H_ * D_ * 2);       // 4 MB
  bf16*  Labb = (bf16*)alloc((size_t)L_ * LD_ * 2);      // 8 MB
  bf16*  Leb  = (bf16*)alloc((size_t)L_ * D_ * 2);       // 8 MB
  float* Lw1  = (float*)alloc((size_t)L_ * H_ * 4);      // 32 MB
  float* Pe   = (float*)alloc((size_t)B_ * D_ * 4);
  float* Pw1  = (float*)alloc((size_t)B_ * H_ * 4);

  // bf16 conversions (label tower + weights)
  cvt_bf16<<<512, 256, 0, stream>>>(W2,  W2b,  H_, H_,  H_,    0);
  cvt_bf16<<<256, 256, 0, stream>>>(Wl,  Wlb,  D_, LD_, LD_,   0);
  cvt_bf16<<<256, 256, 0, stream>>>(W1,  W1lb, H_, D_,  2*D_,  D_);  // W1l slice
  cvt_bf16<<<512, 256, 0, stream>>>(lab, Labb, L_, LD_, LD_,   0);

  // tiny P path in f32
  pe_kernel <<<(B_ * D_ + 255) / 256, 256, 0, stream>>>(seq, Wp, Pe);
  pw1_kernel<<<(B_ * H_ + 255) / 256, 256, 0, stream>>>(Pe, W1, b1, Pw1);

  // L_e = lab @ Wl.T  (M=4000, N=1024, K=1024) -> bf16
  gemm_bf16_nt<true><<<dim3(LD_ / 512, L_ / 16), 256, 0, stream>>>(
      Labb, Wlb, nullptr, Leb, D_, LD_);
  // Lw1 = L_e @ W1l.T (M=4000, N=2048, K=1024) -> f32
  gemm_bf16_nt<false><<<dim3(H_ / 512, L_ / 16), 256, 0, stream>>>(
      Leb, W1lb, Lw1, nullptr, H_, D_);

  // fused h1/h2/logit stage: 128 KB dynamic LDS per block
  (void)hipFuncSetAttribute((const void*)fused_out,
                            hipFuncAttributeMaxDynamicSharedMemorySize,
                            32 * H_ * 2);
  fused_out<<<1000, 256, 32 * H_ * 2, stream>>>(Pw1, Lw1, W2b, b2, Wout, bout, out);
}
